// SearchLayer_12678743458465
// MI455X (gfx1250) — matrix operand order repaired and checked by reading, not compiled, not run on hardware
//
#include <hip/hip_runtime.h>

// ---------------------------------------------------------------------------
// Problem constants (from reference): B=1024, D=512, N=131072, K=16
// ---------------------------------------------------------------------------
#define DDIM   512
#define NCOLS  131072
#define BROWS  1024
#define KTOP   16
#define NSLICES 64
#define SLICE_W (NCOLS / NSLICES)   // 2048 columns per slice
#define NTILES  (SLICE_W / 16)      // 128 n-tiles per slice
#define ROWH    520                 // padded LDS row: 512 bf16 + 8 (16B pad)

typedef __attribute__((ext_vector_type(16))) __bf16 v16bf;
typedef __attribute__((ext_vector_type(8)))  __bf16 v8bf;
typedef __attribute__((ext_vector_type(8)))  float  v8f;

__device__ __forceinline__ unsigned short f2bf(float f) {
  unsigned u = __float_as_uint(f);
  unsigned r = u + 0x7fffu + ((u >> 16) & 1u);   // round-to-nearest-even
  return (unsigned short)(r >> 16);
}

// async global -> LDS, 16 bytes per lane (ASYNCcnt-tracked)
__device__ __forceinline__ void async_cp16(unsigned lds_off, const void* gaddr) {
  asm volatile("global_load_async_to_lds_b128 %0, %1, off"
               :: "v"(lds_off), "v"(gaddr) : "memory");
}
__device__ __forceinline__ void wait_async_le(void) {   // tolerate 4 in flight
  asm volatile("s_wait_asynccnt 0x4" ::: "memory");
}
__device__ __forceinline__ void wait_async_0(void) {
  asm volatile("s_wait_asynccnt 0x0" ::: "memory");
}

// ---------------------------------------------------------------------------
// Kernel A: L2-normalize queries, emit bf16 qn [B, D]
// ---------------------------------------------------------------------------
__global__ __launch_bounds__(256)
void normalize_kernel(const float* __restrict__ q, unsigned short* __restrict__ qn) {
  const int w = threadIdx.x >> 5, lane = threadIdx.x & 31;
  const int row = blockIdx.x * 8 + w;
  const float4* src = (const float4*)(q + (size_t)row * DDIM);
  float4 v[4];
  float ss = 0.f;
#pragma unroll
  for (int j = 0; j < 4; ++j) {
    v[j] = src[j * 32 + lane];
    ss += v[j].x * v[j].x + v[j].y * v[j].y + v[j].z * v[j].z + v[j].w * v[j].w;
  }
#pragma unroll
  for (int off = 16; off >= 1; off >>= 1) ss += __shfl_xor(ss, off, 32);
  const float rn = rsqrtf(ss);
  uint2* dst = (uint2*)(qn + (size_t)row * DDIM);
#pragma unroll
  for (int j = 0; j < 4; ++j) {
    uint2 p;
    p.x = (unsigned)f2bf(v[j].x * rn) | ((unsigned)f2bf(v[j].y * rn) << 16);
    p.y = (unsigned)f2bf(v[j].z * rn) | ((unsigned)f2bf(v[j].w * rn) << 16);
    dst[j * 32 + lane] = p;
  }
}

// ---------------------------------------------------------------------------
// Kernel B: transpose + convert weight [D, N] f32 -> Wt [N, D] bf16
// ---------------------------------------------------------------------------
__global__ __launch_bounds__(256)
void transpose_convert_kernel(const float* __restrict__ W, unsigned short* __restrict__ Wt) {
  __shared__ float tile[32][33];
  const int nb = blockIdx.x * 32;   // N base
  const int db = blockIdx.y * 32;   // D base
  const int t = threadIdx.x;
#pragma unroll
  for (int i = 0; i < 4; ++i) {
    int e = t + i * 256, rr = e >> 5, cc = e & 31;           // rr: d-local, cc: n-local
    tile[rr][cc] = W[(size_t)(db + rr) * NCOLS + nb + cc];
  }
  __syncthreads();
#pragma unroll
  for (int i = 0; i < 4; ++i) {
    int e = t + i * 256, rr = e >> 5, cc = e & 31;           // rr: n-local, cc: d-local
    Wt[(size_t)(nb + rr) * DDIM + db + cc] = f2bf(tile[cc][rr]);
  }
}

// ---------------------------------------------------------------------------
// Kernel C: fused bf16 WMMA GEMM + per-slice running top-16
// B tiles (16 cols x 512 K = 16KB) async-copied global->LDS once per WG,
// double-buffered on ASYNCcnt, shared by 8 waves. A tile register-resident.
// B fragments: depth-2 ds_load pipeline + sched_barrier so wmma waits at
// dscnt<=2 instead of 0.
// ---------------------------------------------------------------------------
__global__ __launch_bounds__(256, 1)
void gemm_topk_kernel(const unsigned short* __restrict__ qn_u,
                      const unsigned short* __restrict__ Wt_u,
                      float* __restrict__ pScore, unsigned* __restrict__ pIdx) {
  __shared__ float    ts[8][16][KTOP];       // per-wave top-K scores
  __shared__ unsigned ti[8][16][KTOP];       // per-wave top-K indices
  __shared__ float    stile[8][16][17];      // per-wave 16x16 sims tile
  __shared__ __align__(16) unsigned short bbuf[2][16][ROWH]; // double-buffered B tile

  const int w    = threadIdx.x >> 5;
  const int lane = threadIdx.x & 31;
  const int t    = threadIdx.x;
  const int slice = blockIdx.x;
  const int mb = (blockIdx.y * 8 + w) * 16;
  const int n0base = slice * SLICE_W;

  const int r = lane & 15;                   // M row (A) / N col (B,C)
  const int h = lane >> 4;                   // half selector

  const __bf16* qb = reinterpret_cast<const __bf16*>(qn_u);
  const __bf16* wb = reinterpret_cast<const __bf16*>(Wt_u);

  // --- issue async copy of tile 0 (4 x b128 per lane-instr, ASYNCcnt += 4) ---
  {
    const char* g0 = (const char*)(wb + (size_t)n0base * DDIM);
#pragma unroll
    for (int i = 0; i < 4; ++i) {
      int e = (t + i * 256) * 16;
      int row = e >> 10, col = e & 1023;
      unsigned loff = (unsigned)(uintptr_t)(&bbuf[0][row][0]) + (unsigned)col;
      async_cp16(loff, g0 + row * (DDIM * 2) + col);
    }
  }

  if (lane < 16) {
#pragma unroll
    for (int k = 0; k < KTOP; ++k) { ts[w][lane][k] = -__builtin_inff(); ti[w][lane][k] = 0; }
  }
  float curMin = -__builtin_inff();
  int   minSlot = 0;

  // --- preload whole 16x512 A tile into registers (16 fragments) ---
  const __bf16* aRow = qb + (size_t)(mb + r) * DDIM;
  v16bf Afrag[16];
#pragma unroll
  for (int s = 0; s < 16; ++s) {
    const __bf16* ap = aRow + s * 32;
    v8bf a0 = *(const v8bf*)(ap + 8 * h);
    v8bf a1 = *(const v8bf*)(ap + 16 + 8 * h);
    Afrag[s] = __builtin_shufflevector(a0, a1, 0,1,2,3,4,5,6,7,8,9,10,11,12,13,14,15);
  }

  for (int nt = 0; nt < NTILES; ++nt) {
    const int cur = nt & 1;
    const int n0 = n0base + nt * 16;

    // issue async copy of next tile into the other buffer
    if (nt + 1 < NTILES) {
      const char* gn = (const char*)(wb + (size_t)(n0 + 16) * DDIM);
#pragma unroll
      for (int i = 0; i < 4; ++i) {
        int e = (t + i * 256) * 16;
        int row = e >> 10, col = e & 1023;
        unsigned loff = (unsigned)(uintptr_t)(&bbuf[1 - cur][row][0]) + (unsigned)col;
        async_cp16(loff, gn + row * (DDIM * 2) + col);
      }
      wait_async_le();          // current tile's 4 ops complete (in-order)
    } else {
      wait_async_0();
    }
    __syncthreads();            // current buffer visible to all waves

    // --- 16 wmma steps; depth-2 pipelined B fragment loads from LDS ---
    const unsigned short* frag = &bbuf[cur][0][0] + r * ROWH + 8 * h;
    v8f acc = {};
    v8bf x0 = *(const v8bf*)(frag);
    v8bf x1 = *(const v8bf*)(frag + 16);
#pragma unroll
    for (int s = 0; s < 16; ++s) {
      v8bf y0, y1;
      if (s + 1 < 16) {
        y0 = *(const v8bf*)(frag + (s + 1) * 32);
        y1 = *(const v8bf*)(frag + (s + 1) * 32 + 16);
      }
      // keep the next-step loads above the wmma (no sinking across this point)
      __builtin_amdgcn_sched_barrier(0);
      v16bf Bm = __builtin_shufflevector(x0, x1,
                                         0,1,2,3,4,5,6,7,8,9,10,11,12,13,14,15);
      acc = __builtin_amdgcn_wmma_f32_16x16x32_bf16(false, Afrag[s], false, Bm,
                                                    (short)0, acc, false, false);
      x0 = y0; x1 = y1;   // rotation (pure renaming under full unroll)
    }

    // C layout: lane holds column n=r, rows v + 8h -> scatter to row-major tile
#pragma unroll
    for (int v = 0; v < 8; ++v) stile[w][v + 8 * h][r] = acc[v];

    // lanes 0..15: insert this tile's candidates (threshold vs cached min)
    if (lane < 16) {
      for (int j = 0; j < 16; ++j) {
        float s = stile[w][lane][j];
        if (s > curMin) {
          ts[w][lane][minSlot] = s;
          ti[w][lane][minSlot] = (unsigned)(n0 + j);
          float mn = ts[w][lane][0]; int msl = 0;
#pragma unroll
          for (int t2 = 1; t2 < KTOP; ++t2) {
            float x = ts[w][lane][t2];
            if (x < mn) { mn = x; msl = t2; }
          }
          curMin = mn; minSlot = msl;
        }
      }
    }
    __syncthreads();            // all reads of buffer done before it is refilled
  }

  // dump per-slice partial lists: pScore[row][slice][k]
  if (lane < 16) {
    const size_t base = (size_t)(mb + lane) * (NSLICES * KTOP) + slice * KTOP;
#pragma unroll
    for (int k = 0; k < KTOP; ++k) {
      pScore[base + k] = ts[w][lane][k];
      pIdx[base + k]   = ti[w][lane][k];
    }
  }
}

// ---------------------------------------------------------------------------
// Kernel D: merge 64 partial top-16 lists per row -> final top-16
// ---------------------------------------------------------------------------
__global__ __launch_bounds__(256)
void merge_topk_kernel(const float* __restrict__ pScore, const unsigned* __restrict__ pIdx,
                       const float* __restrict__ label,
                       float* __restrict__ outScores, float* __restrict__ outIdxF,
                       float* __restrict__ outLabels, unsigned* __restrict__ finalIdx) {
  const int w = threadIdx.x >> 5, lane = threadIdx.x & 31;
  const int b = blockIdx.x * 8 + w;
  const float*    ps = pScore + (size_t)b * (NSLICES * KTOP);
  const unsigned* pi = pIdx   + (size_t)b * (NSLICES * KTOP);
  int sel[KTOP];
#pragma unroll
  for (int round = 0; round < KTOP; ++round) {
    float bestv = -__builtin_inff();
    int bestPos = 0x7fffffff;
    for (int j = 0; j < 32; ++j) {
      int c = lane + j * 32;
      bool used = false;
#pragma unroll
      for (int t = 0; t < KTOP; ++t) used = used || (t < round && sel[t] == c);
      float s = used ? -__builtin_inff() : ps[c];
      if (s > bestv) { bestv = s; bestPos = c; }
    }
#pragma unroll
    for (int off = 16; off >= 1; off >>= 1) {
      float ov = __shfl_xor(bestv, off, 32);
      int   op = __shfl_xor(bestPos, off, 32);
      if (ov > bestv || (ov == bestv && op < bestPos)) { bestv = ov; bestPos = op; }
    }
    sel[round] = bestPos;
    if (lane == 0) {
      unsigned widx = pi[bestPos];
      outScores[b * KTOP + round] = bestv;
      outIdxF  [b * KTOP + round] = (float)widx;
      outLabels[b * KTOP + round] = label[widx];
      finalIdx [b * KTOP + round] = widx;
    }
  }
}

// ---------------------------------------------------------------------------
// Kernel E: gather selected database rows from Wt (contiguous, coalesced)
// ---------------------------------------------------------------------------
__global__ __launch_bounds__(256)
void gather_seq_kernel(const unsigned short* __restrict__ Wt,
                       const unsigned* __restrict__ finalIdx,
                       float* __restrict__ outSeqs) {
  const int p = blockIdx.x;
  const unsigned idx = finalIdx[p];
  const unsigned* src = (const unsigned*)(Wt + (size_t)idx * DDIM);
  const int t = threadIdx.x;
  unsigned v = src[t];
  float2 o;
  o.x = __uint_as_float(v << 16);          // low bf16 -> f32
  o.y = __uint_as_float(v & 0xffff0000u);  // high bf16 -> f32
  *(float2*)(outSeqs + (size_t)p * DDIM + 2 * t) = o;
}

// ---------------------------------------------------------------------------
// Host launch
// ---------------------------------------------------------------------------
extern "C" void kernel_launch(void* const* d_in, const int* in_sizes, int n_in,
                              void* d_out, int out_size, void* d_ws, size_t ws_size,
                              hipStream_t stream) {
  const float* queries = (const float*)d_in[0];   // [1024, 512]
  const float* weight  = (const float*)d_in[1];   // [512, 131072]
  const float* label   = (const float*)d_in[2];   // [131072]

  float* out = (float*)d_out;
  float* outScores = out;                                   // [1024,16]
  float* outIdxF   = out + BROWS * KTOP;                    // [1024,16]
  float* outSeqs   = out + 2 * BROWS * KTOP;                // [16384,512]
  float* outLabels = out + 2 * BROWS * KTOP + (size_t)BROWS * KTOP * DDIM;

  // workspace layout (~144 MB)
  char* ws = (char*)d_ws;
  unsigned short* qn = (unsigned short*)ws;                                  // 1 MB
  unsigned short* Wt = (unsigned short*)(ws + (1u << 20));                   // 134 MB
  char* p0 = ws + (1u << 20) + (size_t)NCOLS * DDIM * 2;
  float*    pScore   = (float*)p0;                                           // 4 MB
  unsigned* pIdx     = (unsigned*)(p0 + (size_t)BROWS * NSLICES * KTOP * 4); // 4 MB
  unsigned* finalIdx = (unsigned*)(p0 + 2 * (size_t)BROWS * NSLICES * KTOP * 4);

  normalize_kernel<<<BROWS / 8, 256, 0, stream>>>(queries, qn);
  transpose_convert_kernel<<<dim3(NCOLS / 32, DDIM / 32), 256, 0, stream>>>(weight, Wt);
  gemm_topk_kernel<<<dim3(NSLICES, BROWS / 128), 256, 0, stream>>>(qn, Wt, pScore, pIdx);
  merge_topk_kernel<<<BROWS / 8, 256, 0, stream>>>(pScore, pIdx, label,
                                                   outScores, outIdxF, outLabels, finalIdx);
  gather_seq_kernel<<<BROWS * KTOP, 256, 0, stream>>>(Wt, finalIdx, outSeqs);
}
